// CTCEnhanced_72456098283992
// MI455X (gfx1250) — compile-verified, matrix operand order checked by reading, not run on hardware
//
#include <hip/hip_runtime.h>
#include <hip/hip_bf16.h>

// ---------------------------------------------------------------------------
// Transformer decoder forward (CTC greedy collapse + 6 decoder layers) for
// MI455X / gfx1250.  All GEMMs run through v_wmma_f32_16x16x32_bf16.
// Weights are converted fp32->bf16 ONCE per launch (L2-resident), Q/K/V are
// produced directly in bf16, so the WMMA inner loops are loads + wmma with
// hardware v_cvt only on the fp32 activation operand.
// ---------------------------------------------------------------------------

#define D_    256
#define SQ    768     // decoder sequence length (== x.shape[0])
#define BB    8       // batch
#define HH    8       // heads
#define HD    32      // head dim
#define VV    80      // vocab
#define TT    760     // y time steps
#define NLAY  6
#define MROWS (SQ*BB) // 6144 token rows

typedef __attribute__((ext_vector_type(16))) __bf16          v16bf;
typedef __attribute__((ext_vector_type(16))) unsigned short  v16us;
typedef __attribute__((ext_vector_type(8)))  unsigned short  v8us;
typedef __attribute__((ext_vector_type(8)))  float           v8f;
typedef unsigned short ushort_t;

static __device__ __forceinline__ ushort_t bf16bits(float f) {
  __bf16 h = (__bf16)f;                       // hardware v_cvt to bf16
  return __builtin_bit_cast(ushort_t, h);
}

static __device__ __forceinline__ v8f wmma_bf16(v16bf a, v16bf b, v8f c) {
  return __builtin_amdgcn_wmma_f32_16x16x32_bf16(
      /*neg_a=*/false, a, /*neg_b=*/false, b,
      /*c_mod=*/(short)0, c, /*reuse_a=*/false, /*reuse_b=*/false);
}

// A/B operand tile fetch from a row-major fp32 [rows,K] row; kb = k0 + (lane<16?0:8).
// Per-lane K pattern (ISA 7.12.2): {kb..kb+7, kb+16..kb+23}.  Hardware cvt to bf16.
static __device__ __forceinline__ v16bf load16_f32(const float* row, int kb) {
  const float4* p0 = (const float4*)(row + kb);
  float4 a = p0[0], b = p0[1];
  const float4* p1 = (const float4*)(row + kb + 16);
  float4 c = p1[0], d = p1[1];
  v16bf t;
  t[0]=(__bf16)a.x;  t[1]=(__bf16)a.y;  t[2]=(__bf16)a.z;  t[3]=(__bf16)a.w;
  t[4]=(__bf16)b.x;  t[5]=(__bf16)b.y;  t[6]=(__bf16)b.z;  t[7]=(__bf16)b.w;
  t[8]=(__bf16)c.x;  t[9]=(__bf16)c.y;  t[10]=(__bf16)c.z; t[11]=(__bf16)c.w;
  t[12]=(__bf16)d.x; t[13]=(__bf16)d.y; t[14]=(__bf16)d.z; t[15]=(__bf16)d.w;
  return t;
}

// Same tile fetch from a pre-converted bf16 [rows,K] row: two 16B loads, no cvt.
static __device__ __forceinline__ v16bf load16_bf16(const ushort_t* row, int kb) {
  v8us a = *(const v8us*)(row + kb);
  v8us b = *(const v8us*)(row + kb + 16);
  v16us t;
#pragma unroll
  for (int i = 0; i < 8; ++i) { t[i] = a[i]; t[8 + i] = b[i]; }
  return __builtin_bit_cast(v16bf, t);
}

// ---------------------------------------------------------------------------
// fp32 -> bf16 weight conversion (once per launch)
// ---------------------------------------------------------------------------
__global__ void wconv_kernel(const float* __restrict__ src, ushort_t* __restrict__ dst, int n) {
  int i = blockIdx.x * 256 + threadIdx.x;
  if (i < n) dst[i] = bf16bits(src[i]);
}

// ---------------------------------------------------------------------------
// CTC greedy collapse: ids[768,8] from argmax(y) with run-length + blank filter
// ---------------------------------------------------------------------------
__global__ void ctc_ids_kernel(const float* __restrict__ y, int* __restrict__ ids) {
  int t = threadIdx.x;
  for (int i = t; i < SQ * BB; i += blockDim.x) ids[i] = VV - 1;  // blank
  __syncthreads();
  if (t < BB) {
    int b = t;
    int prev = -1, rank = 0;
    for (int step = 0; step < TT; ++step) {
      const float* p = y + ((size_t)step * BB + b) * VV;
      float best = p[0]; int bi = 0;
      for (int v = 1; v < VV; ++v) { float f = p[v]; if (f > best) { best = f; bi = v; } }
      bool keep = (bi != prev) && (bi != VV - 1);
      prev = bi;
      if (keep) { ++rank; if (rank < SQ) ids[rank * BB + b] = bi; }
    }
    ids[b] = 0;  // BOS row forced to 0
  }
}

// ---------------------------------------------------------------------------
// Embedding + sinusoidal positional encoding -> h [6144, 256]
// ---------------------------------------------------------------------------
__global__ void embed_kernel(const float* __restrict__ emb, const int* __restrict__ ids,
                             float* __restrict__ h) {
  int row = blockIdx.x, t = threadIdx.x;
  int s = row >> 3;                       // row = s*B + b
  int id = ids[row];
  float df = __expf(-(float)(t & ~1) * (9.210340371976184f / 256.0f)); // ln(1e4)/D
  float ang = (float)s * df;
  float pe = (t & 1) ? __cosf(ang) : __sinf(ang);
  h[(size_t)row * D_ + t] = emb[(size_t)id * D_ + t] + pe;
}

// ---------------------------------------------------------------------------
// WMMA GEMM:  C[M,N] = A[M,K](fp32) * Wb[N,K](bf16)^T + bias, optional relu.
// Each wave computes NT adjacent 16x16 N-tiles sharing the A operand regs.
// mode 0: fp32 row-major [M,N] store.  mode 1: bf16 scatter to [b,h,s,hd].
// ---------------------------------------------------------------------------
template <int NT>
__global__ void gemm_bias_kernel(const float* __restrict__ A, const ushort_t* __restrict__ Wb,
                                 const float* __restrict__ bias, float* __restrict__ Cf,
                                 ushort_t* __restrict__ Cb, int M, int N, int K,
                                 int mode, int relu) {
  int wave   = blockIdx.x * (blockDim.x >> 5) + (threadIdx.x >> 5);
  int groupsN = (N >> 4) / NT;
  int total   = (M >> 4) * groupsN;
  if (wave >= total) return;                 // wave-uniform exit (EXEC all-1 for WMMA)
  int mt = wave / groupsN, ng = wave % groupsN;
  int lane = threadIdx.x & 31;
  int lm   = lane & 15;
  int kof  = (lane & 16) ? 8 : 0;
  int hi8  = (lane & 16) ? 8 : 0;
  const float* arow = A + (size_t)(mt * 16 + lm) * K;
  const ushort_t* wrow[NT];
#pragma unroll
  for (int j = 0; j < NT; ++j)
    wrow[j] = Wb + (size_t)((ng * NT + j) * 16 + lm) * K;
  v8f acc[NT];
#pragma unroll
  for (int j = 0; j < NT; ++j) acc[j] = (v8f){0.f,0.f,0.f,0.f,0.f,0.f,0.f,0.f};
  for (int k0 = 0; k0 < K; k0 += 32) {
    v16bf a = load16_f32(arow, k0 + kof);
#pragma unroll
    for (int j = 0; j < NT; ++j) {
      v16bf b = load16_bf16(wrow[j], k0 + kof);
      acc[j] = wmma_bf16(a, b, acc[j]);
    }
  }
#pragma unroll
  for (int j = 0; j < NT; ++j) {
    int nn = (ng * NT + j) * 16 + lm;
    float bvv = bias ? bias[nn] : 0.f;
#pragma unroll
    for (int r = 0; r < 8; ++r) {
      int mm = mt * 16 + hi8 + r;
      float v = acc[j][r] + bvv;
      if (relu) v = fmaxf(v, 0.f);
      if (mode == 0) {
        Cf[(size_t)mm * N + nn] = v;
      } else {
        int s = mm >> 3, b = mm & 7;            // mm = s*B + b
        int hh = nn >> 5, dd = nn & 31;         // nn = h*32 + d
        Cb[((size_t)((b * HH + hh) * SQ + s)) * HD + dd] = bf16bits(v);
      }
    }
  }
}

// ---------------------------------------------------------------------------
// Flash attention: one wave per (b,h, 16-row q-tile).  Q/K/V bf16 in [b,h,s,32].
// 32 keys per chunk: 2 WMMAs for scores, online softmax, P through LDS to
// A-layout, 2 WMMAs for P*V.  Output fp32 -> [s*B+b, h*32+d].
// ---------------------------------------------------------------------------
__global__ void flash_attn_kernel(const ushort_t* __restrict__ Q, const ushort_t* __restrict__ Kk,
                                  const ushort_t* __restrict__ Vv, float* __restrict__ O,
                                  int causal) {
  __shared__ ushort_t Pl[16 * 32];            // one wave per block
  int bh = blockIdx.x / (SQ / 16);
  int qt = blockIdx.x % (SQ / 16);
  int lane = threadIdx.x & 31;
  int lm   = lane & 15;
  int kof  = (lane & 16) ? 8 : 0;
  int hi8  = (lane & 16) ? 8 : 0;
  const ushort_t* qbp = Q  + (size_t)bh * SQ * HD;
  const ushort_t* kbp = Kk + (size_t)bh * SQ * HD;
  const ushort_t* vbp = Vv + (size_t)bh * SQ * HD;
  int q0 = qt * 16;

  v16bf aq = load16_bf16(qbp + (size_t)(q0 + lm) * HD, kof);

  float mrow[8], lrow[8];
  v8f o0 = {0.f,0.f,0.f,0.f,0.f,0.f,0.f,0.f};
  v8f o1 = {0.f,0.f,0.f,0.f,0.f,0.f,0.f,0.f};
#pragma unroll
  for (int r = 0; r < 8; ++r) { mrow[r] = -1e30f; lrow[r] = 0.f; }

  const float scale = 0.17677669529663687f;   // 1/sqrt(32)
  int kend = causal ? (q0 + 16) : SQ;

  for (int k0 = 0; k0 < kend; k0 += 32) {
    v16bf bk0 = load16_bf16(kbp + (size_t)(k0 + lm)      * HD, kof);
    v16bf bk1 = load16_bf16(kbp + (size_t)(k0 + 16 + lm) * HD, kof);
    v8f z = {0.f,0.f,0.f,0.f,0.f,0.f,0.f,0.f};
    v8f s0 = wmma_bf16(aq, bk0, z);
    v8f s1 = wmma_bf16(aq, bk1, z);
    int key0 = k0 + lm, key1 = k0 + 16 + lm;

    float p0[8], p1[8];
#pragma unroll
    for (int r = 0; r < 8; ++r) {
      float sa = s0[r] * scale, sb = s1[r] * scale;
      if (causal) {
        int q = q0 + hi8 + r;
        if (key0 > q) sa = -1e30f;
        if (key1 > q) sb = -1e30f;
      }
      float cm = fmaxf(sa, sb);
      for (int off = 1; off < 16; off <<= 1) cm = fmaxf(cm, __shfl_xor(cm, off, 32));
      float nm = fmaxf(mrow[r], cm);
      float alpha = __expf(mrow[r] - nm);
      float pa = __expf(sa - nm);
      float pb = __expf(sb - nm);
      float ps = pa + pb;
      for (int off = 1; off < 16; off <<= 1) ps += __shfl_xor(ps, off, 32);
      lrow[r] = lrow[r] * alpha + ps;
      o0[r] *= alpha; o1[r] *= alpha;
      mrow[r] = nm;
      p0[r] = pa; p1[r] = pb;
    }

    // C-layout P -> LDS [16 rows][32 keys] (bf16)
#pragma unroll
    for (int r = 0; r < 8; ++r) {
      Pl[(hi8 + r) * 32 + lm]      = bf16bits(p0[r]);
      Pl[(hi8 + r) * 32 + 16 + lm] = bf16bits(p1[r]);
    }
    __syncthreads();

    // reload P in A-matrix layout (m = lm, K pattern over the 32 keys)
    v16us pt;
#pragma unroll
    for (int j = 0; j < 8; ++j) pt[j]     = Pl[lm * 32 + kof + j];
#pragma unroll
    for (int j = 0; j < 8; ++j) pt[8 + j] = Pl[lm * 32 + kof + 16 + j];
    v16bf pA = __builtin_bit_cast(v16bf, pt);

    // V chunk as two B tiles (N = head-dim halves), strided bf16 loads
    v16us fv0, fv1;
#pragma unroll
    for (int j = 0; j < 8; ++j) {
      int kk  = k0 + kof + j;
      int kk2 = k0 + kof + 16 + j;
      fv0[j]     = vbp[(size_t)kk  * HD + lm];
      fv1[j]     = vbp[(size_t)kk  * HD + 16 + lm];
      fv0[8 + j] = vbp[(size_t)kk2 * HD + lm];
      fv1[8 + j] = vbp[(size_t)kk2 * HD + 16 + lm];
    }
    v16bf bv0 = __builtin_bit_cast(v16bf, fv0);
    v16bf bv1 = __builtin_bit_cast(v16bf, fv1);
    o0 = wmma_bf16(pA, bv0, o0);
    o1 = wmma_bf16(pA, bv1, o1);
    __syncthreads();   // protect Pl against next iteration's stores
  }

  int b = bh >> 3, h = bh & 7;
#pragma unroll
  for (int r = 0; r < 8; ++r) {
    float inv = 1.f / lrow[r];
    int q = q0 + hi8 + r;
    float* dst = O + ((size_t)(q * BB + b)) * D_ + h * HD;
    dst[lm]      = o0[r] * inv;
    dst[16 + lm] = o1[r] * inv;
  }
}

// ---------------------------------------------------------------------------
// LayerNorm(X + R) * g + b   (R may be null; out may alias X)
// ---------------------------------------------------------------------------
__global__ void ln_kernel(const float* __restrict__ X, const float* __restrict__ R,
                          const float* __restrict__ g, const float* __restrict__ b,
                          float* __restrict__ out) {
  __shared__ float red[256];
  int row = blockIdx.x, t = threadIdx.x;
  float v = X[(size_t)row * D_ + t] + (R ? R[(size_t)row * D_ + t] : 0.f);
  red[t] = v; __syncthreads();
  for (int s = 128; s > 0; s >>= 1) { if (t < s) red[t] += red[t + s]; __syncthreads(); }
  float mean = red[0] * (1.f / D_);
  __syncthreads();
  float d = v - mean;
  red[t] = d * d; __syncthreads();
  for (int s = 128; s > 0; s >>= 1) { if (t < s) red[t] += red[t + s]; __syncthreads(); }
  float var = red[0] * (1.f / D_);
  out[(size_t)row * D_ + t] = d * rsqrtf(var + 1e-5f) * g[t] + b[t];
}

// ---------------------------------------------------------------------------
// Host-side orchestration.
// Input order assumption: d_in[0]=x [768,8,256], d_in[1]=y [760,8,80], then
// params flattened in jax tree-leaves order (dict keys alphabetical at each
// level):  emb, {layer: b1,b2, ca(bk,bo,bq,bv,wk,wo,wq,wv), ln1(b,g),
// ln2(b,g), ln3(b,g), sa(bk,bo,bq,bv,wk,wo,wq,wv), w1, w2} x6,
// lnf(b,g), out_b, out_w.   (163 inputs total)
// ---------------------------------------------------------------------------
extern "C" void kernel_launch(void* const* d_in, const int* in_sizes, int n_in,
                              void* d_out, int out_size, void* d_ws, size_t ws_size,
                              hipStream_t stream) {
  (void)in_sizes; (void)n_in; (void)out_size; (void)ws_size;
  const float* x = (const float*)d_in[0];
  const float* y = (const float*)d_in[1];
  const float* const* P = (const float* const*)(d_in + 2);

  const float* emb   = P[0];
  const float* lnf_b = P[1 + 26 * NLAY];
  const float* lnf_g = P[2 + 26 * NLAY];
  const float* out_b = P[3 + 26 * NLAY];
  const float* out_w = P[4 + 26 * NLAY];

  // ---- workspace carve-up (float units) ----
  int*   ids = (int*)d_ws;
  float* w   = (float*)d_ws;
  size_t U   = (size_t)MROWS * D_;          // 1,572,864
  float* h   = w + 8192;
  float* t1  = h  + U;
  float* ao  = t1 + U;
  float* f1  = ao + U;                      // MROWS x 1024
  ushort_t* qb   = (ushort_t*)(f1 + (size_t)MROWS * 1024);  // bf16 [b,h,s,hd]
  ushort_t* kbuf = qb + U;
  ushort_t* vbuf = kbuf + U;
  ushort_t* wpool = vbuf + U;               // bf16 weight pool

  // per-layer bf16 weight pool layout (ushort offsets):
  //   sa_wq, sa_wk, sa_wv, sa_wo, ca_wq, ca_wk, ca_wv, ca_wo  (each 256*256)
  //   w1 (1024*256), w2 (256*1024); then out_w (80*256) after all layers
  const size_t WSQ = (size_t)D_ * D_;       // 65536
  const size_t WFF = (size_t)1024 * D_;     // 262144
  const size_t LBLK = 8 * WSQ + 2 * WFF;    // 1,048,576 per layer

  auto conv = [&](const float* src, ushort_t* dst, int n) {
    wconv_kernel<<<(n + 255) / 256, 256, 0, stream>>>(src, dst, n);
  };
  auto gemm4 = [&](const float* A, const ushort_t* Wb, const float* bias,
                   float* Cf, ushort_t* Cb, int M, int N, int K, int mode, int relu) {
    int groups = (M / 16) * ((N / 16) / 4);
    gemm_bias_kernel<4><<<(groups + 7) / 8, 256, 0, stream>>>(A, Wb, bias, Cf, Cb, M, N, K, mode, relu);
  };
  auto gemm5 = [&](const float* A, const ushort_t* Wb, const float* bias,
                   float* Cf, int M, int N, int K) {
    int groups = (M / 16) * ((N / 16) / 5);
    gemm_bias_kernel<5><<<(groups + 7) / 8, 256, 0, stream>>>(A, Wb, bias, Cf, nullptr, M, N, K, 0, 0);
  };

  // ---- one-time weight conversion ----
  for (int l = 0; l < NLAY; ++l) {
    const float* const* Lp = P + 1 + 26 * l;
    ushort_t* base = wpool + (size_t)l * LBLK;
    conv(Lp[22], base + 0 * WSQ, (int)WSQ);  // sa_wq
    conv(Lp[20], base + 1 * WSQ, (int)WSQ);  // sa_wk
    conv(Lp[23], base + 2 * WSQ, (int)WSQ);  // sa_wv
    conv(Lp[21], base + 3 * WSQ, (int)WSQ);  // sa_wo
    conv(Lp[8],  base + 4 * WSQ, (int)WSQ);  // ca_wq
    conv(Lp[6],  base + 5 * WSQ, (int)WSQ);  // ca_wk
    conv(Lp[9],  base + 6 * WSQ, (int)WSQ);  // ca_wv
    conv(Lp[7],  base + 7 * WSQ, (int)WSQ);  // ca_wo
    conv(Lp[24], base + 8 * WSQ, (int)WFF);  // w1
    conv(Lp[25], base + 8 * WSQ + WFF, (int)WFF);  // w2
  }
  ushort_t* outw_b16 = wpool + (size_t)NLAY * LBLK;
  conv(out_w, outw_b16, VV * D_);

  ctc_ids_kernel<<<1, 256, 0, stream>>>(y, ids);
  embed_kernel<<<MROWS, 256, 0, stream>>>(emb, ids, h);

  const int FLASH_BLOCKS = BB * HH * (SQ / 16);   // 3072 waves

  for (int l = 0; l < NLAY; ++l) {
    const float* const* Lp = P + 1 + 26 * l;
    const float *b1 = Lp[0],    *b2 = Lp[1];
    const float *ca_bk = Lp[2], *ca_bo = Lp[3], *ca_bq = Lp[4], *ca_bv = Lp[5];
    const float *ln1b = Lp[10], *ln1g = Lp[11];
    const float *ln2b = Lp[12], *ln2g = Lp[13];
    const float *ln3b = Lp[14], *ln3g = Lp[15];
    const float *sa_bk = Lp[16], *sa_bo = Lp[17], *sa_bq = Lp[18], *sa_bv = Lp[19];
    ushort_t* base = wpool + (size_t)l * LBLK;
    const ushort_t *sa_wq = base + 0*WSQ, *sa_wk = base + 1*WSQ, *sa_wv = base + 2*WSQ, *sa_wo = base + 3*WSQ;
    const ushort_t *ca_wq = base + 4*WSQ, *ca_wk = base + 5*WSQ, *ca_wv = base + 6*WSQ, *ca_wo = base + 7*WSQ;
    const ushort_t *w1 = base + 8*WSQ, *w2 = base + 8*WSQ + WFF;

    // --- self attention (causal) ---
    gemm4(h, sa_wq, sa_bq, nullptr, qb,   MROWS, D_, D_, 1, 0);
    gemm4(h, sa_wk, sa_bk, nullptr, kbuf, MROWS, D_, D_, 1, 0);
    gemm4(h, sa_wv, sa_bv, nullptr, vbuf, MROWS, D_, D_, 1, 0);
    flash_attn_kernel<<<FLASH_BLOCKS, 32, 0, stream>>>(qb, kbuf, vbuf, ao, 1);
    gemm4(ao, sa_wo, sa_bo, t1, nullptr, MROWS, D_, D_, 0, 0);
    ln_kernel<<<MROWS, 256, 0, stream>>>(h, t1, ln1g, ln1b, h);

    // --- cross attention (keys/values from x) ---
    gemm4(h, ca_wq, ca_bq, nullptr, qb,   MROWS, D_, D_, 1, 0);
    gemm4(x, ca_wk, ca_bk, nullptr, kbuf, MROWS, D_, D_, 1, 0);
    gemm4(x, ca_wv, ca_bv, nullptr, vbuf, MROWS, D_, D_, 1, 0);
    flash_attn_kernel<<<FLASH_BLOCKS, 32, 0, stream>>>(qb, kbuf, vbuf, ao, 0);
    gemm4(ao, ca_wo, ca_bo, t1, nullptr, MROWS, D_, D_, 0, 0);
    ln_kernel<<<MROWS, 256, 0, stream>>>(h, t1, ln2g, ln2b, h);

    // --- feed-forward ---
    gemm4(h,  w1, b1, f1, nullptr, MROWS, 1024, D_,   0, 1);
    gemm4(f1, w2, b2, t1, nullptr, MROWS, D_,   1024, 0, 0);
    ln_kernel<<<MROWS, 256, 0, stream>>>(h, t1, ln3g, ln3b, h);
  }

  ln_kernel<<<MROWS, 256, 0, stream>>>(h, nullptr, lnf_g, lnf_b, h);
  gemm5(h, outw_b16, out_b, (float*)d_out, MROWS, VV, D_);
}
